// GFCNF_83863531422354
// MI455X (gfx1250) — matrix operand order repaired and checked by reading, not compiled, not run on hardware
//
#include <hip/hip_runtime.h>
#include <math.h>

typedef __attribute__((ext_vector_type(16))) __bf16 v16bf;
typedef __attribute__((ext_vector_type(8)))  float  v8f;

// ---------------------------------------------------------------------------
// Per-edge B-spline basis (degree 1, dim 2): 4 (weight, kernel-index) pairs.
// ---------------------------------------------------------------------------
__global__ void basis_kernel(const float* __restrict__ ea, float* __restrict__ basis,
                             int* __restrict__ wi, int K, int E) {
    int e = blockIdx.x * blockDim.x + threadIdx.x;
    if (e >= E) return;
    float vx = ea[2 * e + 0] * (float)(K - 1);
    float vy = ea[2 * e + 1] * (float)(K - 1);
    float fx = floorf(vx), fy = floorf(vy);
    int   kx = (int)fx,    ky = (int)fy;
    float frx = vx - fx,   fry = vy - fy;
#pragma unroll
    for (int s = 0; s < 4; ++s) {
        int b0 = s & 1, b1 = s >> 1;
        float w0 = b0 ? frx : 1.f - frx;
        float w1 = b1 ? fry : 1.f - fry;
        int ix = (kx + b0) % K, iy = (ky + b1) % K;
        basis[4 * e + s] = w0 * w1;
        wi[4 * e + s]    = ix + iy * K;
    }
}

__global__ void deg_kernel(const int* __restrict__ dst, float* __restrict__ deg, int E) {
    int e = blockIdx.x * blockDim.x + threadIdx.x;
    if (e >= E) return;
    atomicAdd(&deg[dst[e]], 1.0f);
}

// Copy x[N,1] into zero-padded [N,32] so conv1a's root GEMM runs guard-free.
__global__ void xpad_kernel(const float* __restrict__ x, float* __restrict__ xp, int Nn) {
    int n = blockIdx.x * blockDim.x + threadIdx.x;
    if (n < Nn) xp[(size_t)n * 32] = x[n];
}

// ---------------------------------------------------------------------------
// Pack f32 weights [Ktot, Cout] -> bf16, zero-padded, B-fragment layout:
//   out[((k>>5)*CoutPad + co)*32 + (k&31)]
// so each WMMA lane's 16 B values are one contiguous 32-byte load.
// ---------------------------------------------------------------------------
__global__ void pack_kernel(const float* __restrict__ W, __bf16* __restrict__ out,
                            int Ktot, int Cout, int Kpad, int CoutPad) {
    int i = blockIdx.x * blockDim.x + threadIdx.x;
    if (i >= Kpad * CoutPad) return;
    int k = i / CoutPad, co = i % CoutPad;
    float v = (k < Ktot && co < Cout) ? W[(size_t)k * Cout + co] : 0.f;
    out[(((size_t)(k >> 5) * CoutPad + co) << 5) + (k & 31)] = (__bf16)v;
}

// ---------------------------------------------------------------------------
// Edge scatter: agg[dst, wi[s]*Cin + c] += basis[s] * x[src, c]
// One wave per edge; lanes stripe channels. K2==1: plain neighbor sum.
// ---------------------------------------------------------------------------
__global__ void scatter_kernel(const float* __restrict__ x, const int* __restrict__ src,
                               const int* __restrict__ dst, const float* __restrict__ basis,
                               const int* __restrict__ wi, float* __restrict__ agg,
                               int Cin, int CinStride, int K2, int Kpad, int E) {
    int edge = blockIdx.x * (blockDim.x >> 5) + (threadIdx.x >> 5);
    if (edge >= E) return;
    int lane = threadIdx.x & 31;
    const float* xs = x + (size_t)src[edge] * CinStride;
    float* ag = agg + (size_t)dst[edge] * Kpad;
    if (K2 == 1) {
        for (int c = lane; c < Cin; c += 32)
            atomicAdd(&ag[c], xs[c]);
    } else {
        float w[4]; int idx[4];
#pragma unroll
        for (int s = 0; s < 4; ++s) { w[s] = basis[4 * edge + s]; idx[s] = wi[4 * edge + s]; }
        for (int c = lane; c < Cin; c += 32) {
            float xv = xs[c];
#pragma unroll
            for (int s = 0; s < 4; ++s)
                atomicAdd(&ag[idx[s] * Cin + c], w[s] * xv);
        }
    }
}

// ---------------------------------------------------------------------------
// Fused SplineConv GEMM: one wave32 computes a 16x32 output tile.
// All MAC-loop loads unconditional & vectorized (padded buffers).
//   accE = agg[16 x Kpad]   @ Wp  (edge part, /deg in epilogue)
//   accR = hin[16 x Cinpad] @ Wrp (root part)
//   out  = elu?(accE/deg + accR + b) (+ residual)
// ---------------------------------------------------------------------------
__device__ __forceinline__ v16bf load_a_frag(const float* p) {
    // a[0..15] = p[0..15]  (contiguous k order, matches packed B)
    float4 u0 = *(const float4*)(p);
    float4 u1 = *(const float4*)(p + 4);
    float4 u2 = *(const float4*)(p + 8);
    float4 u3 = *(const float4*)(p + 12);
    v16bf a;
    a[0]=(__bf16)u0.x; a[1]=(__bf16)u0.y; a[2]=(__bf16)u0.z; a[3]=(__bf16)u0.w;
    a[4]=(__bf16)u1.x; a[5]=(__bf16)u1.y; a[6]=(__bf16)u1.z; a[7]=(__bf16)u1.w;
    a[8]=(__bf16)u2.x; a[9]=(__bf16)u2.y; a[10]=(__bf16)u2.z; a[11]=(__bf16)u2.w;
    a[12]=(__bf16)u3.x; a[13]=(__bf16)u3.y; a[14]=(__bf16)u3.z; a[15]=(__bf16)u3.w;
    return a;
}

__global__ void __launch_bounds__(32)
spline_gemm(const float* __restrict__ agg, const __bf16* __restrict__ Wp,
            const float* __restrict__ hin, const __bf16* __restrict__ Wrp,
            const float* __restrict__ bias, const float* __restrict__ deg,
            const float* __restrict__ residual, float* __restrict__ out,
            int Kpad, int Cinpad, int Cout, int CoutPad, int Nn, int doElu) {
    const int lane   = threadIdx.x & 31;
    const int half16 = lane >> 4;
    const int lidx   = lane & 15;
    const int rowB   = blockIdx.x * 16;
    const int colB   = blockIdx.y * 32;

    const float* aP  = agg + (size_t)(rowB + lidx) * Kpad  + half16 * 16;
    const float* arP = hin + (size_t)(rowB + lidx) * Cinpad + half16 * 16;

    v8f accE0 = {0.f,0.f,0.f,0.f,0.f,0.f,0.f,0.f};
    v8f accE1 = accE0, accR0 = accE0, accR1 = accE0;

    for (int kb = 0; kb < Kpad; kb += 32) {
        v16bf a = load_a_frag(aP + kb);
        const __bf16* bp = Wp + (((size_t)(kb >> 5) * CoutPad + colB + lidx) << 5) + half16 * 16;
        v16bf b0 = *(const v16bf*)(bp);
        v16bf b1 = *(const v16bf*)(bp + 512);   // +16 columns * 32
        accE0 = __builtin_amdgcn_wmma_f32_16x16x32_bf16(false, a, false, b0, (short)0, accE0, false, false);
        accE1 = __builtin_amdgcn_wmma_f32_16x16x32_bf16(false, a, false, b1, (short)0, accE1, false, false);
    }
    for (int kb = 0; kb < Cinpad; kb += 32) {
        v16bf a = load_a_frag(arP + kb);
        const __bf16* bp = Wrp + (((size_t)(kb >> 5) * CoutPad + colB + lidx) << 5) + half16 * 16;
        v16bf b0 = *(const v16bf*)(bp);
        v16bf b1 = *(const v16bf*)(bp + 512);
        accR0 = __builtin_amdgcn_wmma_f32_16x16x32_bf16(false, a, false, b0, (short)0, accR0, false, false);
        accR1 = __builtin_amdgcn_wmma_f32_16x16x32_bf16(false, a, false, b1, (short)0, accR1, false, false);
    }

    float dinv[8];
#pragma unroll
    for (int r = 0; r < 8; ++r) {
        float d = deg[rowB + half16 * 8 + r];
        dinv[r] = 1.f / (d > 1.f ? d : 1.f);
    }
#pragma unroll
    for (int t = 0; t < 2; ++t) {
        int co = colB + t * 16 + lidx;
        if (co >= Cout) continue;
        float bv = bias[co];
        const v8f& aE = t ? accE1 : accE0;
        const v8f& aR = t ? accR1 : accR0;
#pragma unroll
        for (int r = 0; r < 8; ++r) {       // C/D layout: M = half16*8 + r, N = lidx
            int m = rowB + half16 * 8 + r;
            float v = aE[r] * dinv[r] + aR[r] + bv;
            if (doElu) v = v > 0.f ? v : (expf(v) - 1.f);
            if (residual) v += residual[(size_t)m * Cout + co];
            out[(size_t)m * Cout + co] = v;
        }
    }
}

// ---------------------------------------------------------------------------
// BatchNorm over nodes: one block per channel.
// ---------------------------------------------------------------------------
__global__ void bn_kernel(const float* __restrict__ in, float* __restrict__ out,
                          const float* __restrict__ gamma, const float* __restrict__ beta,
                          int C, int Nn) {
    int c = blockIdx.x, t = threadIdx.x;
    __shared__ float s1[256], s2[256];
    float a = 0.f, b = 0.f;
    for (int n = t; n < Nn; n += 256) {
        float v = in[(size_t)n * C + c];
        a += v; b += v * v;
    }
    s1[t] = a; s2[t] = b; __syncthreads();
    for (int o = 128; o > 0; o >>= 1) {
        if (t < o) { s1[t] += s1[t + o]; s2[t] += s2[t + o]; }
        __syncthreads();
    }
    float mu  = s1[0] / (float)Nn;
    float var = s2[0] / (float)Nn - mu * mu;
    float scale = gamma[c] * rsqrtf(var + 1e-5f);
    float shift = beta[c] - mu * scale;
    for (int n = t; n < Nn; n += 256)
        out[(size_t)n * C + c] = in[(size_t)n * C + c] * scale + shift;
}

// ---------------------------------------------------------------------------
extern "C" void kernel_launch(void* const* d_in, const int* in_sizes, int n_in,
                              void* d_out, int out_size, void* d_ws, size_t ws_size,
                              hipStream_t stream) {
    const int Nn = in_sizes[0];          // 65536
    const int E  = in_sizes[1] / 2;      // 524288

    const float* x   = (const float*)d_in[0];
    const int*   src = (const int*)d_in[1];
    const int*   dst = src + E;
    const float* ea  = (const float*)d_in[2];

    auto convP = [&](int i, int which) -> const float* { return (const float*)d_in[3 + 3 * i + which]; };
    static const int convK[15]    = {5,5,3,3,3,1,1,1,1,3,3,1,1,1,5};
    static const int convCin[15]  = {1,32,32,64,64,128,128,256,256,64,128,32,32,32,32};
    static const int convCout[15] = {32,32,64,64,128,128,256,256,32,32,32,32,32,32,1};

    int maxKpad = 0;
    for (int i = 0; i < 15; ++i) {
        int kp = (convK[i] * convK[i] * convCin[i] + 31) & ~31;
        if (kp > maxKpad) maxKpad = kp;
    }

    char* base = (char*)d_ws; size_t off = 0;
    auto alloc = [&](size_t bytes) -> void* {
        void* p = base + off; off += (bytes + 255) & ~(size_t)255; return p;
    };
    float*  deg    = (float*)alloc((size_t)Nn * 4);
    float*  basis3 = (float*)alloc((size_t)E * 4 * 4);
    int*    wi3    = (int*)  alloc((size_t)E * 4 * 4);
    float*  basis5 = (float*)alloc((size_t)E * 4 * 4);
    int*    wi5    = (int*)  alloc((size_t)E * 4 * 4);
    float*  agg    = (float*)alloc((size_t)Nn * maxKpad * 4);
    float*  hA     = (float*)alloc((size_t)Nn * 256 * 4);
    float*  hB     = (float*)alloc((size_t)Nn * 256 * 4);
    float*  pool2  = (float*)alloc((size_t)Nn * 64 * 4);
    float*  pool3  = (float*)alloc((size_t)Nn * 128 * 4);
    float*  t0     = (float*)alloc((size_t)Nn * 32 * 4);
    float*  t1     = (float*)alloc((size_t)Nn * 32 * 4);
    float*  xpad   = (float*)alloc((size_t)Nn * 32 * 4);
    __bf16* wp     = (__bf16*)alloc((size_t)maxKpad * 256 * 2);
    __bf16* wrp    = (__bf16*)alloc((size_t)256 * 256 * 2);

    // graph-wide precompute (reused by every conv)
    hipMemsetAsync(deg, 0, (size_t)Nn * 4, stream);
    deg_kernel<<<(E + 255) / 256, 256, 0, stream>>>(dst, deg, E);
    basis_kernel<<<(E + 255) / 256, 256, 0, stream>>>(ea, basis3, wi3, 3, E);
    basis_kernel<<<(E + 255) / 256, 256, 0, stream>>>(ea, basis5, wi5, 5, E);
    hipMemsetAsync(xpad, 0, (size_t)Nn * 32 * 4, stream);
    xpad_kernel<<<(Nn + 255) / 256, 256, 0, stream>>>(x, xpad, Nn);

    auto run_conv = [&](int ci, const float* hin, int CinStride, float* outBuf,
                        const float* residual, bool elu) {
        int K = convK[ci], Cin = convCin[ci], Cout = convCout[ci];
        int K2 = K * K, Ktot = K2 * Cin;
        int Kpad    = (Ktot + 31) & ~31;
        int CoutPad = (Cout + 31) & ~31;
        hipMemsetAsync(agg, 0, (size_t)Nn * Kpad * 4, stream);
        const float* bas = (K == 5) ? basis5 : basis3;
        const int*   wiP = (K == 5) ? wi5 : wi3;
        scatter_kernel<<<(E + 7) / 8, 256, 0, stream>>>(hin, src, dst, bas, wiP, agg,
                                                        Cin, CinStride, K2, Kpad, E);
        pack_kernel<<<(Kpad * CoutPad + 255) / 256, 256, 0, stream>>>(
            convP(ci, 0), wp, Ktot, Cout, Kpad, CoutPad);
        pack_kernel<<<(CinStride * CoutPad + 255) / 256, 256, 0, stream>>>(
            convP(ci, 1), wrp, Cin, Cout, CinStride, CoutPad);
        dim3 grid(Nn / 16, CoutPad / 32);
        spline_gemm<<<grid, 32, 0, stream>>>(agg, wp, hin, wrp, convP(ci, 2), deg,
                                             residual, outBuf, Kpad, CinStride,
                                             Cout, CoutPad, Nn, elu ? 1 : 0);
    };
    auto run_bn = [&](int bi, const float* in, float* outb, int C) {
        bn_kernel<<<C, 256, 0, stream>>>(in, outb, (const float*)d_in[48 + 2 * bi],
                                         (const float*)d_in[48 + 2 * bi + 1], C, Nn);
    };

    run_conv(0,  xpad, 32,   hA, nullptr, true);   // conv1a   1->32  K5 (padded input)
    run_conv(1,  hA,   32,   hB, nullptr, true);   // conv1b  32->32  K5
    run_bn(0, hB, hA, 32);                         // bn1
    run_conv(2,  hA,   32,   hB, nullptr, true);   // conv2a  32->64  K3
    run_conv(3,  hB,   64,   hA, nullptr, true);   // conv2b  64->64  K3
    run_bn(1, hA, pool2, 64);                      // bn2 -> pool2
    run_conv(4,  pool2, 64,  hA, nullptr, true);   // conv3a  64->128 K3
    run_conv(5,  hA,   128,  hB, nullptr, true);   // conv3b 128->128 K1
    run_bn(2, hB, pool3, 128);                     // bn3 -> pool3
    run_conv(6,  pool3, 128, hA, nullptr, true);   // conv4a 128->256 K1
    run_conv(7,  hA,   256,  hB, nullptr, true);   // conv4b 256->256 K1
    run_bn(3, hB, hA, 256);                        // bn4
    run_conv(8,  hA,   256,  t0, nullptr, true);   // score_fr     256->32 K1
    run_conv(10, pool3, 128, t1, t0,      true);   // score_pool3 + t0
    run_conv(11, t1,   32,   t0, nullptr, true);   // score_w3
    run_conv(9,  pool2, 64,  t1, t0,      true);   // score_pool2 + t0
    run_conv(12, t1,   32,   t0, nullptr, true);   // score_w2
    run_conv(13, t0,   32,   t1, nullptr, true);   // score_w1
    run_conv(14, t1,   32, (float*)d_out, nullptr, false); // convout 32->1 K5, no ELU
}
